// SequenceOneChannel5_9990093930777
// MI455X (gfx1250) — matrix operand order repaired and checked by reading, not compiled, not run on hardware
//
#include <hip/hip_runtime.h>
#include <hip/hip_bf16.h>

// ---------------------------------------------------------------------------
// 5-layer LSTM (H=51) -> only layers 1..3 are live (output reads h3 only).
// Decomposition: 16 blocks x 16 batch rows, each block runs the full T-step
// recurrence for its slice. Per step, per layer: (16xK)x(Kx204) GEMM done
// with v_wmma_f32_16x16x32_bf16, 13 waves = 13 N-tiles of 16.
// Weights: bf16, stacked+padded in d_ws (L2-resident), B-frags via
// global_load_b128. Activations/h/c in LDS, A-frags via ds_load_b128.
// ---------------------------------------------------------------------------

typedef __attribute__((ext_vector_type(16))) __bf16 v16bf;
typedef __attribute__((ext_vector_type(8)))  float  v8f;

#define HSZ 51
#define G4  204          // 4*H
#define NP  208          // padded gate dim: 13 WMMA N-tiles
#define KP1 64           // padded K, layer 1 (h_own only, 51->64)
#define KP  128          // padded K, layers 2/3 ([h_below@0..50 | h_own@64..114])
#define NW  13
#define NT  416          // 13 waves * 32
#define MB  16           // batch rows per block

// d_ws byte offsets
#define WS_W1   0                        // NP*KP1 bf16 = 26624 B
#define WS_W2   (WS_W1 + NP*KP1*2)       // NP*KP  bf16 = 53248 B
#define WS_W3   (WS_W2 + NP*KP*2)
#define WS_F    (WS_W3 + NP*KP*2)        // floats: bs1,bs2,bs3,wx (208 ea), wl(52)

union Frag { uint4 q[2]; v16bf v; };

// A-matrix fragment (16-bit 16x32 layout): lane m=L%16, half=L/16,
// K pairs {s..s+7} ++ {s+16..s+23}, s = kt*32 + 8*half.
__device__ __forceinline__ v16bf load_a_frag(const __bf16* __restrict__ A, int kpad,
                                             int kt, int row, int half) {
    const __bf16* p = A + row * kpad + kt * 32 + 8 * half;
    Frag f;
    f.q[0] = *(const uint4*)(p);        // k = s .. s+7      (16B, aligned)
    f.q[1] = *(const uint4*)(p + 16);   // k = s+16 .. s+23
    return f.v;
}

// B-matrix fragment (32x16): lane n=L%16, half=L/16,
// K {s..s+7} ++ {s+16..s+23}, s = kt*32 + 8*half.  Stored as W[n][k] row-major.
__device__ __forceinline__ v16bf load_b_frag(const __bf16* __restrict__ W, int kpad,
                                             int kt, int row, int half) {
    const __bf16* p = W + row * kpad + kt * 32 + 8 * half;
    Frag f;
    f.q[0] = *(const uint4*)(p);
    f.q[1] = *(const uint4*)(p + 16);
    return f.v;
}

template <int KT, int KPAD>
__device__ __forceinline__ v8f gemm_tile(const __bf16* __restrict__ A,
                                         const __bf16* __restrict__ W,
                                         int row, int half, int nn) {
    v8f acc = {0.f, 0.f, 0.f, 0.f, 0.f, 0.f, 0.f, 0.f};
#pragma unroll
    for (int kt = 0; kt < KT; ++kt) {
        v16bf a = load_a_frag(A, KPAD, kt, row, half);
        v16bf b = load_b_frag(W, KPAD, kt, nn, half);
        acc = __builtin_amdgcn_wmma_f32_16x16x32_bf16(false, a, false, b,
                                                      (short)0, acc, false, false);
    }
    return acc;
}

__device__ __forceinline__ float sigf(float v)  { return 1.f / (1.f + __expf(-v)); }
__device__ __forceinline__ float tanhf_(float v){ return 1.f - 2.f / (__expf(2.f * v) + 1.f); }

// ---------------------------------------------------------------------------
// Prep: fold biases, convert/stack/pad weights to bf16 into d_ws.
// ---------------------------------------------------------------------------
__global__ void lstm_prep_kernel(const float* __restrict__ Wih1, const float* __restrict__ Whh1,
                                 const float* __restrict__ Wih,  const float* __restrict__ Whh,
                                 const float* __restrict__ bih1, const float* __restrict__ bhh1,
                                 const float* __restrict__ bih,  const float* __restrict__ bhh,
                                 const float* __restrict__ Wl,   const float* __restrict__ bl,
                                 unsigned char* __restrict__ ws) {
    __bf16* W1 = (__bf16*)(ws + WS_W1);
    __bf16* W2 = (__bf16*)(ws + WS_W2);
    __bf16* W3 = (__bf16*)(ws + WS_W3);
    float*  bs1 = (float*)(ws + WS_F);
    float*  bs2 = bs1 + NP;
    float*  bs3 = bs2 + NP;
    float*  wx  = bs3 + NP;
    float*  wl  = wx  + NP;

    int idx = blockIdx.x * blockDim.x + threadIdx.x;
    if (idx >= NP * KP) return;
    int n = idx / KP, k = idx % KP;

    // layers 2/3 stacked weights: [Wih@k0..50 | Whh@k64..114], zero pad.
    float v2 = 0.f, v3 = 0.f;
    if (n < G4) {
        if (k < HSZ) {
            v2 = Wih[0 * G4 * HSZ + n * HSZ + k];
            v3 = Wih[1 * G4 * HSZ + n * HSZ + k];
        } else if (k >= 64 && k < 64 + HSZ) {
            v2 = Whh[0 * G4 * HSZ + n * HSZ + (k - 64)];
            v3 = Whh[1 * G4 * HSZ + n * HSZ + (k - 64)];
        }
    }
    W2[n * KP + k] = (__bf16)v2;
    W3[n * KP + k] = (__bf16)v3;

    if (k < KP1) {
        float v1 = (n < G4 && k < HSZ) ? Whh1[n * HSZ + k] : 0.f;
        W1[n * KP1 + k] = (__bf16)v1;
    }
    if (k == 0) {
        float b1 = 0.f, b2 = 0.f, b3 = 0.f, wv = 0.f;
        if (n < G4) {
            b1 = bih1[n] + bhh1[n];
            b2 = bih[n] + bhh[n];                 // layer index 0
            b3 = bih[G4 + n] + bhh[G4 + n];       // layer index 1
            wv = Wih1[n];                         // (4H,1) input weight column
        }
        bs1[n] = b1; bs2[n] = b2; bs3[n] = b3; wx[n] = wv;
    }
    if (idx < HSZ + 1) wl[idx] = (idx < HSZ) ? Wl[idx] : bl[0];
}

// ---------------------------------------------------------------------------
// Main recurrence kernel: one block = 16 batch rows, full T steps.
// ---------------------------------------------------------------------------
__global__ __launch_bounds__(NT) void lstm_seq_kernel(const float* __restrict__ x,
                                                      const unsigned char* __restrict__ ws,
                                                      float* __restrict__ out, int T) {
    __shared__ __bf16 A1[MB * KP1];       // layer1 A: h1 (prev step), zero-padded
    __shared__ __bf16 A2[MB * KP];        // layer2 A: [h1_cur | h2_prev]
    __shared__ __bf16 A3[MB * KP];        // layer3 A: [h2_cur | h3_prev]
    __shared__ float  gbuf[MB * NP];      // fp32 gates for current layer
    __shared__ float  cb[3][MB * 52];     // cell states (stride 52)
    __shared__ float  h3f[MB * 52];       // fp32 h3 for output dot
    __shared__ float  bsl[3][NP];
    __shared__ float  wxl[NP];
    __shared__ float  wll[HSZ + 1];
    __shared__ float  xs[MB];

    const __bf16* W1 = (const __bf16*)(ws + WS_W1);
    const __bf16* W2 = (const __bf16*)(ws + WS_W2);
    const __bf16* W3 = (const __bf16*)(ws + WS_W3);
    const float*  fb = (const float*)(ws + WS_F);

    const int tid = threadIdx.x;
    // --- init LDS state ---
    for (int i = tid; i < MB * KP1; i += NT) A1[i] = (__bf16)0.f;
    for (int i = tid; i < MB * KP;  i += NT) { A2[i] = (__bf16)0.f; A3[i] = (__bf16)0.f; }
    for (int i = tid; i < MB * 52;  i += NT) { cb[0][i] = 0.f; cb[1][i] = 0.f; cb[2][i] = 0.f; }
    for (int i = tid; i < NP;       i += NT) {
        bsl[0][i] = fb[i]; bsl[1][i] = fb[NP + i]; bsl[2][i] = fb[2 * NP + i];
        wxl[i] = fb[3 * NP + i];
    }
    if (tid < HSZ + 1) wll[tid] = fb[4 * NP + tid];
    __syncthreads();

    const int wave = tid >> 5, lane = tid & 31;
    const int row  = lane & 15, half = lane >> 4;
    const int nn   = wave * 16 + row;     // gate column handled by this lane
    const int b0   = blockIdx.x * MB;

    for (int t = 0; t < T; ++t) {
        if (tid < MB) xs[tid] = x[(b0 + tid) * T + t];
        __syncthreads();

        // ---------------- layer 1 ----------------
        {
            v8f acc = gemm_tile<2, KP1>(A1, W1, row, half, nn);
            const float bsv = bsl[0][nn];
            const float xw  = wxl[nn];
#pragma unroll
            for (int r = 0; r < 8; ++r) {
                int m = r + 8 * half;
                gbuf[m * NP + nn] = acc[r] + bsv + xs[m] * xw;  // rank-1 x-input fold
            }
        }
        __syncthreads();
        for (int p = tid; p < MB * HSZ; p += NT) {
            int m = p / HSZ, j = p - m * HSZ;
            float gi = gbuf[m * NP + j];
            float gf = gbuf[m * NP + j + HSZ];
            float gg = gbuf[m * NP + j + 2 * HSZ];
            float go = gbuf[m * NP + j + 3 * HSZ];
            float cN = sigf(gf) * cb[0][m * 52 + j] + sigf(gi) * tanhf_(gg);
            float hN = sigf(go) * tanhf_(cN);
            cb[0][m * 52 + j] = cN;
            __bf16 hb = (__bf16)hN;
            A1[m * KP1 + j] = hb;      // own recurrence (next step)
            A2[m * KP  + j] = hb;      // feed layer 2 (this step)
        }
        __syncthreads();

        // ---------------- layer 2 ----------------
        {
            v8f acc = gemm_tile<4, KP>(A2, W2, row, half, nn);
            const float bsv = bsl[1][nn];
#pragma unroll
            for (int r = 0; r < 8; ++r)
                gbuf[(r + 8 * half) * NP + nn] = acc[r] + bsv;
        }
        __syncthreads();
        for (int p = tid; p < MB * HSZ; p += NT) {
            int m = p / HSZ, j = p - m * HSZ;
            float gi = gbuf[m * NP + j];
            float gf = gbuf[m * NP + j + HSZ];
            float gg = gbuf[m * NP + j + 2 * HSZ];
            float go = gbuf[m * NP + j + 3 * HSZ];
            float cN = sigf(gf) * cb[1][m * 52 + j] + sigf(gi) * tanhf_(gg);
            float hN = sigf(go) * tanhf_(cN);
            cb[1][m * 52 + j] = cN;
            __bf16 hb = (__bf16)hN;
            A2[m * KP + 64 + j] = hb;  // own recurrence (next step)
            A3[m * KP + j]      = hb;  // feed layer 3 (this step)
        }
        __syncthreads();

        // ---------------- layer 3 ----------------
        {
            v8f acc = gemm_tile<4, KP>(A3, W3, row, half, nn);
            const float bsv = bsl[2][nn];
#pragma unroll
            for (int r = 0; r < 8; ++r)
                gbuf[(r + 8 * half) * NP + nn] = acc[r] + bsv;
        }
        __syncthreads();
        for (int p = tid; p < MB * HSZ; p += NT) {
            int m = p / HSZ, j = p - m * HSZ;
            float gi = gbuf[m * NP + j];
            float gf = gbuf[m * NP + j + HSZ];
            float gg = gbuf[m * NP + j + 2 * HSZ];
            float go = gbuf[m * NP + j + 3 * HSZ];
            float cN = sigf(gf) * cb[2][m * 52 + j] + sigf(gi) * tanhf_(gg);
            float hN = sigf(go) * tanhf_(cN);
            cb[2][m * 52 + j] = cN;
            A3[m * KP + 64 + j] = (__bf16)hN;   // own recurrence (next step)
            h3f[m * 52 + j]     = hN;           // fp32 copy for output linear
        }
        __syncthreads();

        // output linear on h3 (layers 4/5 are dead code)
        if (tid < MB) {
            float s = wll[HSZ];
#pragma unroll 3
            for (int j = 0; j < HSZ; ++j) s += h3f[tid * 52 + j] * wll[j];
            out[(b0 + tid) * T + t] = s;
        }
        // xs/h3f hazards covered by the syncthreads at top of next iteration
    }
}

// ---------------------------------------------------------------------------
extern "C" void kernel_launch(void* const* d_in, const int* in_sizes, int n_in,
                              void* d_out, int out_size, void* d_ws, size_t ws_size,
                              hipStream_t stream) {
    const float* x    = (const float*)d_in[0];
    const float* Wih1 = (const float*)d_in[1];
    const float* Whh1 = (const float*)d_in[2];
    const float* bih1 = (const float*)d_in[3];
    const float* bhh1 = (const float*)d_in[4];
    const float* Wih  = (const float*)d_in[5];
    const float* Whh  = (const float*)d_in[6];
    const float* bih  = (const float*)d_in[7];
    const float* bhh  = (const float*)d_in[8];
    const float* Wl   = (const float*)d_in[9];
    const float* bl   = (const float*)d_in[10];

    const int B = 256;
    const int T = in_sizes[0] / B;

    lstm_prep_kernel<<<(NP * KP + 255) / 256, 256, 0, stream>>>(
        Wih1, Whh1, Wih, Whh, bih1, bhh1, bih, bhh, Wl, bl, (unsigned char*)d_ws);

    lstm_seq_kernel<<<B / MB, NT, 0, stream>>>(
        x, (const unsigned char*)d_ws, (float*)d_out, T);

    (void)n_in; (void)out_size; (void)ws_size;
}
